// Non_local_Block_17867063951542
// MI455X (gfx1250) — matrix operand order repaired and checked by reading, not compiled, not run on hardware
//
#include <hip/hip_runtime.h>
#include <hip/hip_bf16.h>

// ---------------------------------------------------------------------------
// Non-local block (dot-product variant, no softmax) via matrix associativity:
//   y = theta_x @ (phi_x @ g_x) / N      (the [N,N] matrix never exists)
// Heavy GEMMs on V_WMMA_F32_16X16X32_BF16; tile staging uses
// GLOBAL_LOAD_ASYNC_TO_LDS_B128 + s_wait_asynccnt when the toolchain has the
// builtins, with a synchronous fallback. Streamed tiles are double-buffered
// (one barrier per K-step); invariant A panels are staged once in LDS.
// ---------------------------------------------------------------------------

typedef __attribute__((ext_vector_type(16))) __bf16 v16bf;
typedef __attribute__((ext_vector_type(8)))  float  v8f;
typedef __attribute__((ext_vector_type(4)))  int    v4i_;

#define LDK  40     // padded stride for 32-wide K tiles (bf16 elems)
#define LDA1 264    // K1 weight panel stride (256 + 8)
#define LDA4 136    // K4 W2 panel stride (128 + 8)
#define NPIX 4096
#define CIN  256

#if defined(__gfx1250__) && __has_builtin(__builtin_amdgcn_global_load_async_to_lds_b128) && __has_builtin(__builtin_amdgcn_s_wait_asynccnt)
#define USE_ASYNC_LDS 1
#else
#define USE_ASYNC_LDS 0
#endif

union Frag16 { uint4 q[2]; v16bf v; };

static __device__ __forceinline__ unsigned short f32_to_bf16_raw(float f) {
    union { float f; unsigned u; } c; c.f = f;
    unsigned r = c.u + 0x7FFFu + ((c.u >> 16) & 1u);   // round-to-nearest-even
    return (unsigned short)(r >> 16);
}

static __device__ __forceinline__ v8f v8f_zero() {
    v8f z;
#pragma unroll
    for (int e = 0; e < 8; ++e) z[e] = 0.0f;
    return z;
}

// 16B global -> LDS copy; async DMA path when available (bypasses VGPRs)
static __device__ __forceinline__ void copy16_g2l(const unsigned short* g, unsigned short* l) {
#if USE_ASYNC_LDS
    __builtin_amdgcn_global_load_async_to_lds_b128(
        (__attribute__((address_space(1))) v4i_*)g,
        (__attribute__((address_space(3))) v4i_*)l, 0, 0);
#else
    *(uint4*)l = *(const uint4*)g;
#endif
}

static __device__ __forceinline__ void lds_sync() {
#if USE_ASYNC_LDS
    __builtin_amdgcn_s_wait_asynccnt(0);
#endif
    __syncthreads();
}

// A fragment 16x32 (MxK), LDS row-major, stride ld
static __device__ __forceinline__ v16bf load_frag_a(const unsigned short* s, int lane, int ld) {
    int m  = lane & 15;
    int kb = (lane & 16) >> 1;              // 0 or 8
    Frag16 f;
    f.q[0] = *(const uint4*)(s + m * ld + kb);
    f.q[1] = *(const uint4*)(s + m * ld + kb + 16);
    return f.v;
}

// B fragment 32x16 (KxN) from LDS stored [N][K], stride LDK
static __device__ __forceinline__ v16bf load_frag_b(const unsigned short* s, int lane) {
    int n  = lane & 15;
    int kb = (lane & 16) >> 1;              // 0 or 8
    Frag16 f;
    f.q[0] = *(const uint4*)(s + n * LDK + kb);
    f.q[1] = *(const uint4*)(s + n * LDK + kb + 16);
    return f.v;
}

static __device__ __forceinline__ v8f wmma_bf16(v16bf a, v16bf b, v8f c) {
    return __builtin_amdgcn_wmma_f32_16x16x32_bf16(false, a, false, b,
                                                   (short)0, c, false, false);
}

// ---------------------------------------------------------------------------
// k0: pack g_w / theta_w / phi_w (f32 [128,256]) into bf16 Wcat[384,256]
// ---------------------------------------------------------------------------
__global__ void k0_prep(const float* __restrict__ g_w,
                        const float* __restrict__ th_w,
                        const float* __restrict__ ph_w,
                        unsigned short* __restrict__ wcat) {
    int i = blockIdx.x * 256 + threadIdx.x;
    if (i < 384 * 256) {
        int row = i >> 8, col = i & 255;
        int sel = row >> 7, c = row & 127;
        const float* src = (sel == 0) ? g_w : (sel == 1) ? th_w : ph_w;
        wcat[i] = f32_to_bf16_raw(src[c * 256 + col]);
    }
}

// ---------------------------------------------------------------------------
// k1: PGT[b] = Wcat[384,256] @ x[b][256,4096] + bias  (bf16 out)
// rows 0-127 = G, 128-255 = Theta, 256-383 = Phi
// ---------------------------------------------------------------------------
__global__ __launch_bounds__(256) void k1_proj(const float* __restrict__ x,
                                               const unsigned short* __restrict__ wcat,
                                               const float* __restrict__ g_b,
                                               const float* __restrict__ th_b,
                                               const float* __restrict__ ph_b,
                                               unsigned short* __restrict__ pgt) {
    __shared__ unsigned short sA[128 * LDA1];       // full 128x256 weight panel
    __shared__ unsigned short sB[2][128 * LDK];     // double-buffered x tile
    const int nt = blockIdx.x, mt = blockIdx.y, b = blockIdx.z;
    const int tid = threadIdx.x, lane = tid & 31, wave = tid >> 5;
    const int wm = wave >> 1, wn = wave & 1;
    const int m0 = mt * 128, n0 = nt * 128;

    v8f acc[2][4];
#pragma unroll
    for (int i = 0; i < 2; ++i)
#pragma unroll
        for (int j = 0; j < 4; ++j) acc[i][j] = v8f_zero();

    // ---- stage full A panel once (async DMA when available) ----
    {
        const int row = tid >> 1, half = (tid & 1) * 128;
#pragma unroll
        for (int q = 0; q < 8; ++q)
            copy16_g2l(wcat + (size_t)(m0 + row) * 256 + half + q * 16,
                       sA + row * LDA1 + half + q * 16);
    }

    // ---- B tile stager: x[b] f32 [k0..k0+32) x [n0..n0+128) -> bf16 [n][k]
    auto stageB = [&](int kt, unsigned short* buf) {
        const int k0 = kt * 32;
        const int kr = tid >> 5;            // 0..7
        const int nc = (tid & 31) * 4;      // 0..124
#pragma unroll
        for (int p = 0; p < 4; ++p) {
            int k = kr + p * 8;
            const float4 v = *(const float4*)(x + ((size_t)(b * CIN + k0 + k)) * NPIX + n0 + nc);
            buf[(nc + 0) * LDK + k] = f32_to_bf16_raw(v.x);
            buf[(nc + 1) * LDK + k] = f32_to_bf16_raw(v.y);
            buf[(nc + 2) * LDK + k] = f32_to_bf16_raw(v.z);
            buf[(nc + 3) * LDK + k] = f32_to_bf16_raw(v.w);
        }
    };

    stageB(0, sB[0]);
    lds_sync();

    for (int kt = 0; kt < 8; ++kt) {
        const int cur = kt & 1;
        if (kt < 7) stageB(kt + 1, sB[1 - cur]);            // prefetch next
        const unsigned short* aP = sA + kt * 32;
        v16bf af[2], bfr[4];
        af[0] = load_frag_a(aP + (wm * 32 + 0) * LDA1, lane, LDA1);
        af[1] = load_frag_a(aP + (wm * 32 + 16) * LDA1, lane, LDA1);
#pragma unroll
        for (int j = 0; j < 4; ++j) bfr[j] = load_frag_b(sB[cur] + (wn * 64 + j * 16) * LDK, lane);
#pragma unroll
        for (int i = 0; i < 2; ++i)
#pragma unroll
            for (int j = 0; j < 4; ++j) acc[i][j] = wmma_bf16(af[i], bfr[j], acc[i][j]);
        lds_sync();                                          // one barrier/iter
    }

    const float* bias = (mt == 0) ? g_b : (mt == 1) ? th_b : ph_b;
    const int cBase = lane & 15;
    const int rOff = (lane >> 4) * 8;
#pragma unroll
    for (int i = 0; i < 2; ++i)
#pragma unroll
        for (int j = 0; j < 4; ++j)
#pragma unroll
            for (int r = 0; r < 8; ++r) {
                int lrow = wm * 32 + i * 16 + rOff + r;
                int gcol = n0 + wn * 64 + j * 16 + cBase;
                float val = acc[i][j][r] + bias[lrow];
                pgt[((size_t)(b * 384) + m0 + lrow) * NPIX + gcol] = f32_to_bf16_raw(val);
            }
}

// ---------------------------------------------------------------------------
// k2: Mpart[b][ks] = Phi[:, kslice] @ G[:, kslice]^T   (K split 8 ways)
// Phi = PGT rows 256..383, G = PGT rows 0..127; both tiles are plain copies
// -> ideal for the async-to-LDS DMA path. Double buffered.
// ---------------------------------------------------------------------------
__global__ __launch_bounds__(256) void k2_moment(const unsigned short* __restrict__ pgt,
                                                 float* __restrict__ mpart) {
    __shared__ unsigned short sA[2][128 * LDK];
    __shared__ unsigned short sB[2][128 * LDK];
    const int ks = blockIdx.x, b = blockIdx.y;
    const int tid = threadIdx.x, lane = tid & 31, wave = tid >> 5;
    const int wm = wave >> 1, wn = wave & 1;

    v8f acc[2][4];
#pragma unroll
    for (int i = 0; i < 2; ++i)
#pragma unroll
        for (int j = 0; j < 4; ++j) acc[i][j] = v8f_zero();

    const int arow = tid >> 1, ahalf = (tid & 1) * 16;

    auto stage = [&](int kt, int buf) {
        const int k0 = (ks * 16 + kt) * 32;
        const unsigned short* gA = pgt + ((size_t)(b * 384) + 256 + arow) * NPIX + k0 + ahalf;
        const unsigned short* gB = pgt + ((size_t)(b * 384) + arow) * NPIX + k0 + ahalf;
        unsigned short* dA = sA[buf] + arow * LDK + ahalf;
        unsigned short* dB = sB[buf] + arow * LDK + ahalf;
        copy16_g2l(gA, dA);       copy16_g2l(gA + 8, dA + 8);
        copy16_g2l(gB, dB);       copy16_g2l(gB + 8, dB + 8);
    };

    stage(0, 0);
    lds_sync();

    for (int kt = 0; kt < 16; ++kt) {
        const int cur = kt & 1;
        if (kt < 15) stage(kt + 1, 1 - cur);
        v16bf af[2], bfr[4];
        af[0] = load_frag_a(sA[cur] + (wm * 32 + 0) * LDK, lane, LDK);
        af[1] = load_frag_a(sA[cur] + (wm * 32 + 16) * LDK, lane, LDK);
#pragma unroll
        for (int j = 0; j < 4; ++j) bfr[j] = load_frag_b(sB[cur] + (wn * 64 + j * 16) * LDK, lane);
#pragma unroll
        for (int i = 0; i < 2; ++i)
#pragma unroll
            for (int j = 0; j < 4; ++j) acc[i][j] = wmma_bf16(af[i], bfr[j], acc[i][j]);
        lds_sync();
    }

    const int cBase = lane & 15;
    const int rOff = (lane >> 4) * 8;
#pragma unroll
    for (int i = 0; i < 2; ++i)
#pragma unroll
        for (int j = 0; j < 4; ++j)
#pragma unroll
            for (int r = 0; r < 8; ++r) {
                int lrow = wm * 32 + i * 16 + rOff + r;
                int lcol = wn * 64 + j * 16 + cBase;
                mpart[(((size_t)b * 8 + ks) * 128 + lrow) * 128 + lcol] = acc[i][j][r];
            }
}

// ---------------------------------------------------------------------------
// k3: reduce K-split partials, then W2[b] = (1/N) * out_w @ M[b]^T (bf16 out)
// ---------------------------------------------------------------------------
__global__ __launch_bounds__(256) void k3_w2(const float* __restrict__ out_w,
                                             const float* __restrict__ mpart,
                                             unsigned short* __restrict__ w2) {
    __shared__ float sM[128 * 128];
    const int b = blockIdx.x, tid = threadIdx.x;
    for (int i = tid; i < 128 * 128; i += 256) {
        float s = 0.0f;
#pragma unroll
        for (int p = 0; p < 8; ++p) s += mpart[((size_t)b * 8 + p) * 16384 + i];
        sM[i] = s * (1.0f / (float)NPIX);
    }
    __syncthreads();
    for (int o = tid; o < 256 * 128; o += 256) {
        int k = o >> 7, c = o & 127;
        const float* wr = out_w + k * 128;
        const float* mr = sM + c * 128;
        float s = 0.0f;
#pragma unroll 8
        for (int q = 0; q < 128; ++q) s += wr[q] * mr[q];
        w2[((size_t)b * 256 + k) * 128 + c] = f32_to_bf16_raw(s);
    }
}

// ---------------------------------------------------------------------------
// k4: out[b] = x[b] + W2[b] @ Theta[b] + out_b   (Theta = PGT rows 128..255)
// ---------------------------------------------------------------------------
__global__ __launch_bounds__(256) void k4_apply(const float* __restrict__ x,
                                                const unsigned short* __restrict__ w2,
                                                const unsigned short* __restrict__ pgt,
                                                const float* __restrict__ out_b,
                                                float* __restrict__ out) {
    __shared__ unsigned short sA[128 * LDA4];       // full 128x128 W2 panel
    __shared__ unsigned short sB[2][128 * LDK];
    const int nt = blockIdx.x, mt = blockIdx.y, b = blockIdx.z;
    const int tid = threadIdx.x, lane = tid & 31, wave = tid >> 5;
    const int wm = wave >> 1, wn = wave & 1;
    const int m0 = mt * 128, n0 = nt * 128;

    v8f acc[2][4];
#pragma unroll
    for (int i = 0; i < 2; ++i)
#pragma unroll
        for (int j = 0; j < 4; ++j) acc[i][j] = v8f_zero();

    // ---- stage full W2 panel once ----
    {
        const int row = tid >> 1, half = (tid & 1) * 64;
#pragma unroll
        for (int q = 0; q < 4; ++q)
            copy16_g2l(w2 + ((size_t)(b * 256) + m0 + row) * 128 + half + q * 16,
                       sA + row * LDA4 + half + q * 16);
    }

    // ---- B stager: Theta rows k0..k0+31 -> LDS transposed [n][k] ----
    auto stageB = [&](int kt, unsigned short* buf) {
        const int k0 = kt * 32;
        const int kr = tid >> 4;            // 0..15
        const int ng = (tid & 15) * 8;      // 0..120
#pragma unroll
        for (int p = 0; p < 2; ++p) {
            int k = kr + p * 16;
            const uint4 v = *(const uint4*)(pgt + ((size_t)(b * 384) + 128 + k0 + k) * NPIX + n0 + ng);
            const unsigned short* e = (const unsigned short*)&v;
#pragma unroll
            for (int j = 0; j < 8; ++j) buf[(ng + j) * LDK + k] = e[j];
        }
    };

    stageB(0, sB[0]);
    lds_sync();

    for (int kt = 0; kt < 4; ++kt) {
        const int cur = kt & 1;
        if (kt < 3) stageB(kt + 1, sB[1 - cur]);
        const unsigned short* aP = sA + kt * 32;
        v16bf af[2], bfr[4];
        af[0] = load_frag_a(aP + (wm * 32 + 0) * LDA4, lane, LDA4);
        af[1] = load_frag_a(aP + (wm * 32 + 16) * LDA4, lane, LDA4);
#pragma unroll
        for (int j = 0; j < 4; ++j) bfr[j] = load_frag_b(sB[cur] + (wn * 64 + j * 16) * LDK, lane);
#pragma unroll
        for (int i = 0; i < 2; ++i)
#pragma unroll
            for (int j = 0; j < 4; ++j) acc[i][j] = wmma_bf16(af[i], bfr[j], acc[i][j]);
        lds_sync();
    }

    const int cBase = lane & 15;
    const int rOff = (lane >> 4) * 8;
#pragma unroll
    for (int i = 0; i < 2; ++i)
#pragma unroll
        for (int j = 0; j < 4; ++j)
#pragma unroll
            for (int r = 0; r < 8; ++r) {
                int grow = m0 + wm * 32 + i * 16 + rOff + r;
                int gcol = n0 + wn * 64 + j * 16 + cBase;
                size_t idx = ((size_t)(b * CIN) + grow) * NPIX + gcol;
                out[idx] = acc[i][j][r] + x[idx] + out_b[grow];
            }
}

// ---------------------------------------------------------------------------
extern "C" void kernel_launch(void* const* d_in, const int* in_sizes, int n_in,
                              void* d_out, int out_size, void* d_ws, size_t ws_size,
                              hipStream_t stream) {
    const float* x     = (const float*)d_in[0];
    const float* g_w   = (const float*)d_in[1];
    const float* g_b   = (const float*)d_in[2];
    const float* th_w  = (const float*)d_in[3];
    const float* th_b  = (const float*)d_in[4];
    const float* ph_w  = (const float*)d_in[5];
    const float* ph_b  = (const float*)d_in[6];
    const float* out_w = (const float*)d_in[7];
    const float* out_b = (const float*)d_in[8];
    float* out = (float*)d_out;

    char* ws = (char*)d_ws;
    const size_t off_wcat  = 0;                                      // 196608 B
    const size_t off_pgt   = off_wcat + 384 * 256 * 2;               // 12582912 B
    const size_t off_mpart = off_pgt + (size_t)4 * 384 * 4096 * 2;   // 2097152 B
    const size_t off_w2    = off_mpart + (size_t)4 * 8 * 128 * 128 * 4; // 262144 B
    unsigned short* wcat  = (unsigned short*)(ws + off_wcat);
    unsigned short* pgt   = (unsigned short*)(ws + off_pgt);
    float*          mpart = (float*)(ws + off_mpart);
    unsigned short* w2    = (unsigned short*)(ws + off_w2);

    k0_prep  <<<dim3(384),      256, 0, stream>>>(g_w, th_w, ph_w, wcat);
    k1_proj  <<<dim3(32, 3, 4), 256, 0, stream>>>(x, wcat, g_b, th_b, ph_b, pgt);
    k2_moment<<<dim3(8, 4),     256, 0, stream>>>(pgt, mpart);
    k3_w2    <<<dim3(4),        256, 0, stream>>>(out_w, mpart, w2);
    k4_apply <<<dim3(32, 2, 4), 256, 0, stream>>>(x, w2, pgt, out_b, out);
}
